// SparseAttention_80900003987608
// MI455X (gfx1250) — matrix operand order repaired and checked by reading, not compiled, not run on hardware
//
#include <hip/hip_runtime.h>
#include <hip/hip_bf16.h>

typedef __attribute__((ext_vector_type(16))) _Float16 v16h;
typedef __attribute__((ext_vector_type(8)))  float    v8f;

union Frag16 { v16h v; unsigned u[8]; _Float16 h[16]; };

static __device__ __forceinline__ unsigned pack2(_Float16 lo, _Float16 hi) {
    union { _Float16 h[2]; unsigned u; } t;
    t.h[0] = lo; t.h[1] = hi;
    return t.u;
}

// Low 32 bits of a generic pointer to __shared__ = LDS byte offset.
static __device__ __forceinline__ unsigned lds_off(const void* p) {
    return (unsigned)(unsigned long long)(uintptr_t)p;
}

// CDNA5 async copy: 16B per lane, global -> LDS, tracked by ASYNCcnt.
static __device__ __forceinline__ void async_copy_b128(unsigned ldsoff, const void* g) {
    asm volatile("global_load_async_to_lds_b128 %0, %1, off"
                 :: "v"(ldsoff), "v"((unsigned long long)(uintptr_t)g)
                 : "memory");
}
static __device__ __forceinline__ void wait_async0() {
    asm volatile("s_wait_asynccnt 0x0" ::: "memory");
}

// ---------------- f32 -> f16 convert ----------------
__global__ void cvt_f32_to_f16(const float* __restrict__ in,
                               _Float16* __restrict__ out, int n) {
    int i = blockIdx.x * blockDim.x + threadIdx.x;
    if (i < n) out[i] = (_Float16)in[i];
}

// Transposing convert: in R x Cc (row-major) -> out Cc x R (row-major), f32->f16
__global__ void cvt_f32_to_f16_T(const float* __restrict__ in,
                                 _Float16* __restrict__ out, int R, int Cc) {
    int i = blockIdx.x * blockDim.x + threadIdx.x;
    if (i < R * Cc) {
        int r = i / Cc, c = i % Cc;
        out[(size_t)c * R + r] = (_Float16)in[i];
    }
}

// ---------------- tiled WMMA GEMM: Y[M,Nn] = A[M,K] @ Bt[Nn,K]^T ----------------
// A: M x K f16 row-major; Bt: Nn x K f16 row-major (pre-transposed weights).
// OUTH: write f16; else f32 (+ bias if BIAS). 128x128 tile, BK=32, 8 waves,
// double-buffered LDS filled with async b128 copies overlapped with WMMA.
template<bool OUTH, bool BIAS>
__global__ __launch_bounds__(256) void gemm_wmma(
    const _Float16* __restrict__ A, const _Float16* __restrict__ Bt,
    _Float16* __restrict__ outh, float* __restrict__ outf,
    const float* __restrict__ bias, int M, int Nn, int K)
{
    constexpr int BM = 128, BN = 128, BK = 32, LDP = BK + 8; // 80B row stride, 16B-aligned
    __shared__ __align__(16) _Float16 lA[2][BM][LDP];
    __shared__ __align__(16) _Float16 lB[2][BN][LDP];

    const int tid   = threadIdx.x;
    const int lane  = tid & 31;
    const int wave  = tid >> 5;
    const int waveM = (wave >> 1) * 32;   // 4 row-quadrants
    const int waveN = (wave & 1) * 64;    // 2 col-halves
    const int m     = lane & 15;
    const int hv    = lane >> 4;
    const int row0  = blockIdx.y * BM;
    const int col0  = blockIdx.x * BN;

    v8f acc[2][4];
#pragma unroll
    for (int mt = 0; mt < 2; mt++)
#pragma unroll
        for (int nt = 0; nt < 4; nt++)
#pragma unroll
            for (int r = 0; r < 8; r++) acc[mt][nt][r] = 0.0f;

    auto stage = [&](int bufi, int k0) {
#pragma unroll
        for (int i = 0; i < 2; i++) {
            int c  = tid + i * 256;           // 0..511 chunks of 8 f16
            int r  = c >> 2;
            int kc = (c & 3) * 8;
            async_copy_b128(lds_off(&lA[bufi][r][kc]),
                            &A[(size_t)(row0 + r) * K + k0 + kc]);
            async_copy_b128(lds_off(&lB[bufi][r][kc]),
                            &Bt[(size_t)(col0 + r) * K + k0 + kc]);
        }
    };

    stage(0, 0);
    wait_async0();
    __syncthreads();

    int buf = 0;
    for (int k0 = 0; k0 < K; k0 += BK) {
        if (k0 + BK < K) stage(buf ^ 1, k0 + BK);   // prefetch next tile (async)

        Frag16 fa[2], fb[4];
#pragma unroll
        for (int mt = 0; mt < 2; mt++) {
            int mr = waveM + mt * 16 + m;
#pragma unroll
            for (int r = 0; r < 8; r++) {
                int k = ((r >> 2) << 4) + (hv << 3) + ((r & 3) << 1);
                fa[mt].u[r] = *(const unsigned*)&lA[buf][mr][k];
            }
        }
#pragma unroll
        for (int nt = 0; nt < 4; nt++) {
            int nr = waveN + nt * 16 + m;
#pragma unroll
            for (int r = 0; r < 8; r++) {
                int k = ((r >> 2) << 4) + (hv << 3) + ((r & 3) << 1);
                fb[nt].u[r] = *(const unsigned*)&lB[buf][nr][k];
            }
        }
#pragma unroll
        for (int mt = 0; mt < 2; mt++)
#pragma unroll
            for (int nt = 0; nt < 4; nt++)
                acc[mt][nt] = __builtin_amdgcn_wmma_f32_16x16x32_f16(
                    false, fa[mt].v, false, fb[nt].v,
                    (short)0, acc[mt][nt], false, false);

        wait_async0();     // my prefetch landed
        __syncthreads();   // everyone done reading buf & writing buf^1
        buf ^= 1;
    }

#pragma unroll
    for (int mt = 0; mt < 2; mt++)
#pragma unroll
        for (int nt = 0; nt < 4; nt++)
#pragma unroll
            for (int r = 0; r < 8; r++) {
                int row = row0 + waveM + mt * 16 + r + hv * 8;
                int col = col0 + waveN + nt * 16 + m;
                size_t idx = (size_t)row * Nn + col;
                float v = acc[mt][nt][r];
                if (OUTH) outh[idx] = (_Float16)v;
                else      outf[idx] = BIAS ? (v + bias[col]) : v;
            }
}

// ---------------- block-diagonal attention: 8 waves/WG, one (b,h,block) per wave ----
// qkv: f16 (B*N, 2304); aout: f16 flat (B,H,N,D) == reference's reshaped (B,N,C)
__global__ __launch_bounds__(256) void attn_wmma(const _Float16* __restrict__ qkv,
                                                 _Float16* __restrict__ aout)
{
    constexpr int N = 8192, H = 12, D = 64, NB = N / 16;
    constexpr size_t RS = 2304;
    constexpr int LQP = 3 * D + 8;                       // 200 halves = 400B stride
    __shared__ __align__(16) _Float16 lq[8][16][LQP];    // per-wave Q|K|V slab
    __shared__ float sp[8][16][17];                      // per-wave probs transpose

    const int wave = threadIdx.x >> 5;
    const int lane = threadIdx.x & 31;
    const int id   = blockIdx.x * 8 + wave;
    const int blk  = id % NB;
    const int h    = (id / NB) % H;
    const int b    = id / (NB * H);
    const int m    = lane & 15;
    const int hv   = lane >> 4;
    const size_t rowbase = (size_t)b * N + (size_t)blk * 16;

    // ---- async-stage Q|K|V (16 tokens x dims 0..31 of each of Q, K, V) into wave-private LDS
#pragma unroll
    for (int sec = 0; sec < 3; sec++) {
        int gcol = sec * 768 + h * D;
#pragma unroll
        for (int j = 0; j < 2; j++) {
            int c  = lane + j * 32;        // 64 chunks: 16 rows x 4
            int r  = c >> 2;
            int kc = (c & 3) * 8;
            async_copy_b128(lds_off(&lq[wave][r][sec * 64 + kc]),
                            &qkv[(rowbase + r) * RS + gcol + kc]);
        }
    }
    wait_async0();   // wave-private data: no cross-wave barrier needed

    // ---- scores = (Q @ K^T) * 1/sqrt(D)
    v8f s;
#pragma unroll
    for (int r = 0; r < 8; r++) s[r] = 0.0f;
    Frag16 fq, fk;
#pragma unroll
    for (int kk = 0; kk < D; kk += 32) {
#pragma unroll
        for (int r = 0; r < 8; r++) {
            int kd = ((r >> 2) << 4) + (hv << 3) + ((r & 3) << 1) + kk;
            fq.u[r] = *(const unsigned*)&lq[wave][m][kd];
        }
#pragma unroll
        for (int r = 0; r < 8; r++) {
            int kd = ((r >> 2) << 4) + (hv << 3) + ((r & 3) << 1) + kk;
            fk.u[r] = *(const unsigned*)&lq[wave][m][64 + kd];
        }
        s = __builtin_amdgcn_wmma_f32_16x16x32_f16(false, fq.v, false, fk.v,
                                                   (short)0, s, false, false);
    }

    // ---- row softmax across each 16-lane group
    float p[8];
#pragma unroll
    for (int r = 0; r < 8; r++) {
        float v = s[r] * 0.125f;
        float mx = v;
#pragma unroll
        for (int off = 8; off >= 1; off >>= 1)
            mx = fmaxf(mx, __shfl_xor(mx, off));
        float e = __expf(v - mx);
        float sm = e;
#pragma unroll
        for (int off = 8; off >= 1; off >>= 1)
            sm += __shfl_xor(sm, off);
        p[r] = e / sm;
    }

    // ---- probs C-layout -> A-fragment via LDS (K padded 16->32 with zeros)
#pragma unroll
    for (int r = 0; r < 8; r++) sp[wave][r + hv * 8][m] = p[r];
    __syncthreads();
    Frag16 fp;
#pragma unroll
    for (int r = 0; r < 4; r++) {
        int k0 = hv * 8 + r * 2;
        fp.u[r] = pack2((_Float16)sp[wave][m][k0], (_Float16)sp[wave][m][k0 + 1]);
    }
#pragma unroll
    for (int r = 4; r < 8; r++) fp.u[r] = 0u;

    // ---- out = P @ V over 4 head-dim tiles
    const size_t obase = (size_t)b * N * 768 + (size_t)h * N * D;
#pragma unroll
    for (int nt = 0; nt < 4; nt++) {
        Frag16 fv;
#pragma unroll
        for (int r = 0; r < 8; r++) {
            int t0 = hv * 8 + 2 * (r & 3);
            unsigned val = pack2(lq[wave][t0][128 + nt * 16 + m],
                                 lq[wave][t0 + 1][128 + nt * 16 + m]);
            fv.u[r] = (r < 4) ? val : 0u;
        }
        v8f o;
#pragma unroll
        for (int r = 0; r < 8; r++) o[r] = 0.0f;
        o = __builtin_amdgcn_wmma_f32_16x16x32_f16(false, fp.v, false, fv.v,
                                                   (short)0, o, false, false);
#pragma unroll
        for (int r = 0; r < 8; r++) {
            int tok = blk * 16 + r + hv * 8;
            aout[obase + (size_t)tok * D + nt * 16 + m] = (_Float16)o[r];
        }
    }
}

// ---------------- launch ----------------
extern "C" void kernel_launch(void* const* d_in, const int* in_sizes, int n_in,
                              void* d_out, int out_size, void* d_ws, size_t ws_size,
                              hipStream_t stream) {
    const float* x     = (const float*)d_in[0];
    const float* Wqkv  = (const float*)d_in[1];
    const float* Wproj = (const float*)d_in[2];
    const float* bproj = (const float*)d_in[3];
    float* out = (float*)d_out;

    const int Bb = 4, N = 8192, C = 768, H = 12;
    const int M = Bb * N;                          // 32768
    const size_t nX   = (size_t)M * C;             // 25,165,824
    const size_t nWq  = (size_t)C * 3 * C;
    const size_t nWp  = (size_t)C * C;
    const size_t nQKV = (size_t)M * 3 * C;

    char* ws = (char*)d_ws;
    size_t off = 0;
    auto wsalloc = [&](size_t bytes) -> char* {
        char* p = ws + off;
        off += (bytes + 255) & ~(size_t)255;
        return p;
    };
    _Float16* xh   = (_Float16*)wsalloc(nX * 2);
    _Float16* wqt  = (_Float16*)wsalloc(nWq * 2);   // Wqkv^T: (3C) x C
    _Float16* wpt  = (_Float16*)wsalloc(nWp * 2);   // Wproj^T: C x C
    _Float16* qkvh = (_Float16*)wsalloc(nQKV * 2);
    _Float16* ath  = (_Float16*)wsalloc(nX * 2);

    const int T = 256;
    cvt_f32_to_f16<<<(unsigned)((nX + T - 1) / T), T, 0, stream>>>(x, xh, (int)nX);
    cvt_f32_to_f16_T<<<(unsigned)((nWq + T - 1) / T), T, 0, stream>>>(Wqkv, wqt, C, 3 * C);
    cvt_f32_to_f16_T<<<(unsigned)((nWp + T - 1) / T), T, 0, stream>>>(Wproj, wpt, C, C);

    // qkv = x @ Wqkv  (f16 out)
    gemm_wmma<true, false><<<dim3(3 * C / 128, M / 128), 256, 0, stream>>>(
        xh, wqt, qkvh, nullptr, nullptr, M, 3 * C, C);

    // block-diagonal attention (8 blocks per workgroup)
    attn_wmma<<<dim3((unsigned)(Bb * H * (N / 16) / 8)), 256, 0, stream>>>(qkvh, ath);

    // out = attn @ Wproj + bias  (f32 out)
    gemm_wmma<false, true><<<dim3(C / 128, M / 128), 256, 0, stream>>>(
        ath, wpt, nullptr, out, bproj, M, C, C);
}
